// GumbelRNNGenerator_72490458021933
// MI455X (gfx1250) — compile-verified, matrix operand order checked by reading, not run on hardware
//
#include <hip/hip_runtime.h>
#include <hip/hip_bf16.h>
#include <math.h>

// Problem dims (fixed by setup_inputs)
#define Bn    128
#define NOISE 128
#define Hd    512
#define H2    1024
#define H3    1536
#define Vv    5000
#define VPAD  5120   // multiple of 256 (block macro-tile width) -> no guards, EXEC all-1s
#define STEPS 128

typedef __attribute__((ext_vector_type(16))) __bf16 v16bf;
typedef __attribute__((ext_vector_type(8)))  float  v8f;

// ---------------- helpers ----------------

__device__ __forceinline__ unsigned short f32_to_bf16(float f) {
    unsigned int u = __float_as_uint(f);
    u += 0x7FFFu + ((u >> 16) & 1u);   // round-to-nearest-even
    return (unsigned short)(u >> 16);
}

// Load one 16x32 bf16 WMMA fragment from global row-major storage.
// ISA 16-bit A/B layout: lane half=lane/16, element j has K = (j&7) + 16*(j>>3) + 8*half,
// i.e. two contiguous 8-element (16-byte) runs at k0+8*half and k0+8*half+16.
__device__ __forceinline__ v16bf load_frag_g(const unsigned short* __restrict__ p,
                                             int ld, int row, int k0, int half) {
    const unsigned short* q = p + (size_t)row * ld + k0 + (half << 3);
    union { v16bf v; uint4 u[2]; } f;
    f.u[0] = *(const uint4*)(q);
    f.u[1] = *(const uint4*)(q + 16);
    return f.v;
}

__device__ __forceinline__ float sigmoidf_(float x) { return 1.0f / (1.0f + __expf(-x)); }

// ---------------- WMMA GEMM: C[M,Npad] = A[M,K] * Bt[Npad,K]^T (+bias) ----------------
// A, Bt bf16 row-major; C f32 row-major ldc=Npad.
// Block = 128 threads (4 waves). Block macro-tile = 32 rows x 256 cols.
// Each wave: 32x64 (2 M-tiles x 4 N-tiles = 8 WMMAs per K-step of 32).
// The 32x32 A K-slab is shared by all 4 waves: staged in LDS with
// global_load_async_to_lds_b128 (double-buffered, ASYNCcnt + barrier synced).
// Requires: M % 32 == 0, Npad % 256 == 0, K % 32 == 0 (pad rows zero-filled).
__global__ void wmma_gemm_bf16(const unsigned short* __restrict__ A,
                               const unsigned short* __restrict__ Bt,
                               const float* __restrict__ bias,
                               float* __restrict__ C,
                               int K, int Npad, int Nbias) {
    __shared__ unsigned short sA[2][32][32];   // 2 x 2KB double buffer

    const int tid  = threadIdx.x;
    const int wave = tid >> 5;
    const int lane = tid & 31;
    const int half = lane >> 4;
    const int l15  = lane & 15;

    const int mbase = blockIdx.y * 32;
    const int nbase = (blockIdx.x * 4 + wave) * 64;

    // async A staging: each of 128 threads copies one 16B chunk per buffer fill
    const int arow   = tid >> 2;               // 0..31
    const int achunk = (tid & 3) << 3;         // element offset 0,8,16,24
    const unsigned short* agsrc = A + (size_t)(mbase + arow) * K + achunk;
    const unsigned int lds_base = (unsigned int)(size_t)(&sA[0][arow][achunk]);

    v8f acc[2][4];
#pragma unroll
    for (int i = 0; i < 2; ++i)
#pragma unroll
        for (int j = 0; j < 4; ++j) acc[i][j] = (v8f){};

    // preload buffer 0
    {
        const unsigned short* g = agsrc;
        asm volatile("global_load_async_to_lds_b128 %0, %1, off"
                     :: "v"(lds_base), "v"(g) : "memory");
    }

    const int iters = K >> 5;
    for (int it = 0; it < iters; ++it) {
        const int buf = it & 1;
        const int k0  = it << 5;
        if (it + 1 < iters) {   // kick off next slab into the other buffer
            const unsigned int laddr = lds_base + (unsigned int)((buf ^ 1) * 2048);
            const unsigned short* g = agsrc + (k0 + 32);
            asm volatile("global_load_async_to_lds_b128 %0, %1, off"
                         :: "v"(laddr), "v"(g) : "memory");
            asm volatile("s_wait_asynccnt 1" ::: "memory");   // current slab landed
        } else {
            asm volatile("s_wait_asynccnt 0" ::: "memory");
        }
        __syncthreads();        // slab visible to all 4 waves

        // A fragments from LDS (ds_load_b128 x2 per tile)
        v16bf afrag[2];
#pragma unroll
        for (int i = 0; i < 2; ++i) {
            const char* ab = (const char*)&sA[buf][i * 16 + l15][half << 3];
            union { v16bf v; uint4 u[2]; } f;
            f.u[0] = *(const uint4*)(ab);
            f.u[1] = *(const uint4*)(ab + 32);
            afrag[i] = f.v;
        }
        // B fragments from global (L2-resident weights)
        v16bf bfrag[4];
#pragma unroll
        for (int j = 0; j < 4; ++j) {
            const int brow = nbase + j * 16 + l15;
            __builtin_prefetch(Bt + (size_t)brow * K + k0 + 256, 0, 1);
            bfrag[j] = load_frag_g(Bt, K, brow, k0, half);
        }
#pragma unroll
        for (int i = 0; i < 2; ++i)
#pragma unroll
            for (int j = 0; j < 4; ++j)
                acc[i][j] = __builtin_amdgcn_wmma_f32_16x16x32_bf16(
                    false, afrag[i], false, bfrag[j], (short)0, acc[i][j], false, false);

        __syncthreads();        // all reads of buf done before it is refilled
    }

    // D layout: m = vgpr + 8*half, n = lane%16
#pragma unroll
    for (int i = 0; i < 2; ++i) {
        const int m0 = mbase + i * 16 + (half << 3);
#pragma unroll
        for (int j = 0; j < 4; ++j) {
            const int ncol = nbase + j * 16 + l15;
            float bv = 0.0f;
            if (bias != nullptr && ncol < Nbias) bv = bias[ncol];
#pragma unroll
            for (int r = 0; r < 8; ++r)
                C[(size_t)(m0 + r) * Npad + ncol] = acc[i][j][r] + bv;
        }
    }
}

// ---------------- setup kernels ----------------

__global__ void cvt_bf16_pad(const float* __restrict__ in, unsigned short* __restrict__ out,
                             int n_in, int n_out) {
    int i = blockIdx.x * blockDim.x + threadIdx.x;
    if (i < n_out) out[i] = (i < n_in) ? f32_to_bf16(in[i]) : (unsigned short)0;
}

__global__ void lrelu_f32(const float* __restrict__ in, float* __restrict__ out, int n) {
    int i = blockIdx.x * blockDim.x + threadIdx.x;
    if (i < n) { float v = in[i]; out[i] = v > 0.0f ? v : 0.2f * v; }
}

// zh = lrelu(z @ W_z2h.T + b_z2h), tiny GEMM (128x512x128) -> naive f32
__global__ void z2h_kernel(const float* __restrict__ z, const float* __restrict__ W,
                           const float* __restrict__ bias, float* __restrict__ out) {
    int idx = blockIdx.x * blockDim.x + threadIdx.x;
    if (idx >= Bn * Hd) return;
    int b = idx / Hd, j = idx % Hd;
    float acc = bias[j];
    const float* zr = z + (size_t)b * NOISE;
    const float* wr = W + (size_t)j * NOISE;
#pragma unroll 4
    for (int k = 0; k < NOISE; ++k) acc = fmaf(zr[k], wr[k], acc);
    out[idx] = acc > 0.0f ? acc : 0.2f * acc;
}

__global__ void init_state(const float* __restrict__ zbn, float* __restrict__ h,
                           unsigned short* __restrict__ hbf, int* __restrict__ prev) {
    int idx = blockIdx.x * blockDim.x + threadIdx.x;
    if (idx < Bn * Hd) { float v = zbn[idx]; h[idx] = v; hbf[idx] = f32_to_bf16(v); }
    if (idx < Bn) prev[idx] = Vv - 1;   // SOS token
}

// ---------------- per-step kernels ----------------

// BatchNorm over batch dim (128 rows). One 128-thread block per column.
__global__ void bn_cols(const float* __restrict__ x, const float* __restrict__ g,
                        const float* __restrict__ beta, float* __restrict__ out_f32,
                        unsigned short* __restrict__ out_bf16, int C) {
    __shared__ float s1[128];
    __shared__ float s2[128];
    const int col = blockIdx.x;
    const int t   = threadIdx.x;
    float v = x[(size_t)t * C + col];
    s1[t] = v; s2[t] = v * v;
    __syncthreads();
    for (int s = 64; s > 0; s >>= 1) {
        if (t < s) { s1[t] += s1[t + s]; s2[t] += s2[t + s]; }
        __syncthreads();
    }
    float mean = s1[0] * (1.0f / 128.0f);
    float var  = s2[0] * (1.0f / 128.0f) - mean * mean;
    float y = g[col] * (v - mean) * rsqrtf(var + 1e-5f) + beta[col];
    if (out_f32)  out_f32[(size_t)t * C + col] = y;
    if (out_bf16) out_bf16[(size_t)t * C + col] = f32_to_bf16(y);
}

// xcat[b, :] = concat(lrelu(emb[prev[b]]), zbn[b])
__global__ void gather_concat(const float* __restrict__ embL, const float* __restrict__ zbn,
                              const int* __restrict__ prev, float* __restrict__ xcat) {
    int idx = blockIdx.x * blockDim.x + threadIdx.x;
    if (idx >= Bn * H2) return;
    int b = idx / H2, c = idx % H2;
    xcat[idx] = (c < Hd) ? embL[(size_t)prev[b] * Hd + c]
                         : zbn[(size_t)b * Hd + (c - Hd)];
}

// GRU gating: consumes gi[B,3H], gh[B,3H]; updates h, h_bf16, hl=lrelu(h)
__global__ void gru_gate(const float* __restrict__ gi, const float* __restrict__ gh,
                         float* __restrict__ h, unsigned short* __restrict__ hbf,
                         float* __restrict__ hl) {
    int idx = blockIdx.x * blockDim.x + threadIdx.x;
    if (idx >= Bn * Hd) return;
    int b = idx / Hd, j = idx % Hd;
    const float* gir = gi + (size_t)b * H3;
    const float* ghr = gh + (size_t)b * H3;
    float r  = sigmoidf_(gir[j]          + ghr[j]);
    float zt = sigmoidf_(gir[Hd + j]     + ghr[Hd + j]);
    float n  = tanhf    (gir[2 * Hd + j] + r * ghr[2 * Hd + j]);
    float hn = (1.0f - zt) * n + zt * h[idx];
    h[idx]   = hn;
    hbf[idx] = f32_to_bf16(hn);
    hl[idx]  = hn > 0.0f ? hn : 0.2f * hn;
}

// Gumbel-softmax sample + argmax. One 256-thread block per batch row.
__global__ void gumbel_sample(const float* __restrict__ logits, const float* __restrict__ bias,
                              const float* __restrict__ temp, float* __restrict__ out,
                              int* __restrict__ prev, int step) {
    __shared__ float rmax[256];
    __shared__ int   rarg[256];
    __shared__ float rsum[256];
    const int b = blockIdx.x, t = threadIdx.x;
    float* orow = out + ((size_t)b * STEPS + step) * Vv;
    const float* lrow = logits + (size_t)b * VPAD;
    const float invT = 1.0f / temp[0];

    float lm = -1e30f; int la = 0;
    for (int v = t; v < Vv; v += 256) {
        // deterministic counter-based hash RNG -> uniform (0,1)
        unsigned int x = (unsigned int)(step + 1) * 0x9E3779B9u
                       ^ (unsigned int)(b + 1)    * 0x85EBCA6Bu
                       ^ (unsigned int)(v + 1)    * 0xC2B2AE35u;
        x ^= x >> 16; x *= 0x7FEB352Du; x ^= x >> 15; x *= 0x846CA68Bu; x ^= x >> 16;
        float u   = fmaxf((float)(x >> 8) * (1.0f / 16777216.0f), 1e-20f);
        float gum = -logf(-logf(u));
        float s   = (lrow[v] + bias[v] + gum) * invT;
        orow[v] = s;
        if (s > lm) { lm = s; la = v; }
    }
    rmax[t] = lm; rarg[t] = la;
    __syncthreads();
    for (int s = 128; s > 0; s >>= 1) {
        if (t < s && rmax[t + s] > rmax[t]) { rmax[t] = rmax[t + s]; rarg[t] = rarg[t + s]; }
        __syncthreads();
    }
    const float mx = rmax[0];
    float ls = 0.0f;
    for (int v = t; v < Vv; v += 256) {
        float e = __expf(orow[v] - mx);
        orow[v] = e;
        ls += e;
    }
    rsum[t] = ls;
    __syncthreads();
    for (int s = 128; s > 0; s >>= 1) {
        if (t < s) rsum[t] += rsum[t + s];
        __syncthreads();
    }
    const float inv = 1.0f / rsum[0];
    for (int v = t; v < Vv; v += 256) orow[v] *= inv;
    if (t == 0) prev[b] = rarg[0];
}

// ---------------- host launcher ----------------

static inline int cdiv(int a, int b) { return (a + b - 1) / b; }

extern "C" void kernel_launch(void* const* d_in, const int* in_sizes, int n_in,
                              void* d_out, int out_size, void* d_ws, size_t ws_size,
                              hipStream_t stream) {
    const float* z      = (const float*)d_in[0];
    const float* temp   = (const float*)d_in[1];
    // d_in[2] = num_steps (int, fixed 128 per setup)
    const float* W_z2h  = (const float*)d_in[3];
    const float* b_z2h  = (const float*)d_in[4];
    const float* g1     = (const float*)d_in[5];
    const float* be1    = (const float*)d_in[6];
    const float* emb    = (const float*)d_in[7];
    const float* g2     = (const float*)d_in[8];
    const float* be2    = (const float*)d_in[9];
    const float* W_ih   = (const float*)d_in[10];
    const float* b_ih   = (const float*)d_in[11];
    const float* W_hh   = (const float*)d_in[12];
    const float* b_hh   = (const float*)d_in[13];
    const float* g3     = (const float*)d_in[14];
    const float* be3    = (const float*)d_in[15];
    const float* W_h2o  = (const float*)d_in[16];
    const float* b_h2o  = (const float*)d_in[17];
    float* out = (float*)d_out;

    // workspace carve-up (256B aligned)
    char* ws = (char*)d_ws;
    size_t off = 0;
    auto alloc = [&](size_t bytes) -> void* {
        void* p = ws + off;
        off += (bytes + 255) & ~(size_t)255;
        return p;
    };
    unsigned short* wih_bf  = (unsigned short*)alloc((size_t)H3 * H2 * 2);
    unsigned short* whh_bf  = (unsigned short*)alloc((size_t)H3 * Hd * 2);
    unsigned short* wh2o_bf = (unsigned short*)alloc((size_t)VPAD * Hd * 2);
    float*          embL    = (float*)alloc((size_t)Vv * Hd * 4);
    float*          zh      = (float*)alloc((size_t)Bn * Hd * 4);
    float*          zbn     = (float*)alloc((size_t)Bn * Hd * 4);
    float*          h       = (float*)alloc((size_t)Bn * Hd * 4);
    unsigned short* h_bf    = (unsigned short*)alloc((size_t)Bn * Hd * 2);
    float*          xcat    = (float*)alloc((size_t)Bn * H2 * 4);
    unsigned short* si_bf   = (unsigned short*)alloc((size_t)Bn * H2 * 2);
    float*          gi      = (float*)alloc((size_t)Bn * H3 * 4);
    float*          gh      = (float*)alloc((size_t)Bn * H3 * 4);
    float*          hl      = (float*)alloc((size_t)Bn * Hd * 4);
    unsigned short* o_bf    = (unsigned short*)alloc((size_t)Bn * Hd * 2);
    float*          logits  = (float*)alloc((size_t)Bn * VPAD * 4);
    int*            prev    = (int*)alloc((size_t)Bn * 4);
    (void)ws_size; (void)n_in; (void)in_sizes; (void)out_size;

    const int TB = 256;
    // one-time weight conversion / precompute (re-run every call for determinism)
    cvt_bf16_pad<<<cdiv(H3 * H2, TB), TB, 0, stream>>>(W_ih,  wih_bf,  H3 * H2, H3 * H2);
    cvt_bf16_pad<<<cdiv(H3 * Hd, TB), TB, 0, stream>>>(W_hh,  whh_bf,  H3 * Hd, H3 * Hd);
    cvt_bf16_pad<<<cdiv(VPAD * Hd, TB), TB, 0, stream>>>(W_h2o, wh2o_bf, Vv * Hd, VPAD * Hd);
    lrelu_f32<<<cdiv(Vv * Hd, TB), TB, 0, stream>>>(emb, embL, Vv * Hd);
    z2h_kernel<<<cdiv(Bn * Hd, TB), TB, 0, stream>>>(z, W_z2h, b_z2h, zh);
    bn_cols<<<Hd, 128, 0, stream>>>(zh, g1, be1, zbn, nullptr, Hd);
    init_state<<<cdiv(Bn * Hd, TB), TB, 0, stream>>>(zbn, h, h_bf, prev);

    const dim3 gemm_block(128);
    const dim3 grid_gate(H3 / 256, Bn / 32);    // 6 x 4, N=1536
    const dim3 grid_out (VPAD / 256, Bn / 32);  // 20 x 4, N=5120

    for (int step = 0; step < STEPS; ++step) {
        gather_concat<<<cdiv(Bn * H2, TB), TB, 0, stream>>>(embL, zbn, prev, xcat);
        bn_cols<<<H2, 128, 0, stream>>>(xcat, g2, be2, nullptr, si_bf, H2);
        wmma_gemm_bf16<<<grid_gate, gemm_block, 0, stream>>>(si_bf, wih_bf, b_ih, gi, H2, H3, H3);
        wmma_gemm_bf16<<<grid_gate, gemm_block, 0, stream>>>(h_bf,  whh_bf, b_hh, gh, Hd, H3, H3);
        gru_gate<<<cdiv(Bn * Hd, TB), TB, 0, stream>>>(gi, gh, h, h_bf, hl);
        bn_cols<<<Hd, 128, 0, stream>>>(hl, g3, be3, nullptr, o_bf, Hd);
        wmma_gemm_bf16<<<grid_out, gemm_block, 0, stream>>>(o_bf, wh2o_bf, nullptr, logits, Hd, VPAD, 0);
        gumbel_sample<<<Bn, 256, 0, stream>>>(logits, b_h2o, temp, out, prev, step);
    }
}
